// EnhancedGAT_9766755631345
// MI455X (gfx1250) — matrix operand order repaired and checked by reading, not compiled, not run on hardware
//
#include <hip/hip_runtime.h>

// ---------------------------------------------------------------------------
// CDNA5 / gfx1250 EnhancedGAT: 3x GATConv (H=4, Fh=32, HF=128) + fused MLP.
// GEMMs run on v_wmma_f32_16x16x32_bf16 (wave32 fragments, A/B in contiguous
// k = 16*(lane/16) + s order); edge softmax-aggregation uses uint-key atomicMax + f32 atomics.
// ---------------------------------------------------------------------------

typedef __attribute__((ext_vector_type(16))) __bf16 v16bf;
typedef __attribute__((ext_vector_type(8)))  float  v8f;

#define NEG_SLOPE 0.2f
#define EPI_BIAS 1
#define EPI_RELU 2
#define EPI_BN   4

__device__ __forceinline__ __bf16 f2bf(float f) {            // RNE f32->bf16
  unsigned u = __float_as_uint(f);
  unsigned r = (u + 0x7FFFu + ((u >> 16) & 1u)) >> 16;
  unsigned short s = (unsigned short)r;
  return __builtin_bit_cast(__bf16, s);
}
__device__ __forceinline__ unsigned bfbits(float f) {
  return (unsigned)__builtin_bit_cast(unsigned short, f2bf(f));
}
// monotone float <-> uint mapping for atomicMax on signed floats
__device__ __forceinline__ unsigned fkey(float f) {
  unsigned u = __float_as_uint(f);
  return (u & 0x80000000u) ? ~u : (u | 0x80000000u);
}
__device__ __forceinline__ float keyf(unsigned k) {
  unsigned u = (k & 0x80000000u) ? (k & 0x7FFFFFFFu) : ~k;
  return __uint_as_float(u);
}
__device__ __forceinline__ float lrelu(float x) { return x > 0.0f ? x : NEG_SLOPE * x; }

// --------------------------- small utility kernels -------------------------

__global__ void cvt_f32_bf16(const float* __restrict__ src, __bf16* __restrict__ dst, int n) {
  int t = blockIdx.x * blockDim.x + threadIdx.x;
  if (t < n) dst[t] = f2bf(src[t]);
}

// Pack a KxN row-major f32 weight into wave32 WMMA B-fragment order:
// P[((ntile*(K/32)+ks)*32 + lane)*8 + j] = { B[k0][n], B[k0+1][n] } as 2xbf16,
// with n = ntile*16 + (lane&15), k0 = ks*32 + (lane>=16 ? 16 : 0) + 2*j.
__global__ void pack_wmma_b(const float* __restrict__ W, unsigned* __restrict__ P, int K, int N) {
  int t = blockIdx.x * blockDim.x + threadIdx.x;
  int total = (K * N) >> 1;
  if (t >= total) return;
  int j = t & 7;
  int lane = (t >> 3) & 31;
  int rest = t >> 8;
  int ksteps = K >> 5;
  int ks = rest % ksteps;
  int nt = rest / ksteps;
  int n  = nt * 16 + (lane & 15);
  int k0 = ks * 32 + ((lane >> 4) << 4) + 2 * j;
  unsigned lo = bfbits(W[k0 * N + n]);
  unsigned hi = bfbits(W[(k0 + 1) * N + n]);
  P[t] = lo | (hi << 16);
}

// --------------------------- WMMA GEMM (bf16 -> f32 acc) -------------------
// C[M,N] = A[M,K] @ B[K,N]; A row-major bf16, B pre-packed fragments.
// Optional fused epilogue: +bias, ReLU, BatchNorm(eval). Writes f32 and/or bf16.
__global__ __launch_bounds__(128)
void gemm_wmma_bf16(const __bf16* __restrict__ A, const unsigned* __restrict__ Bp,
                    float* __restrict__ Cf, __bf16* __restrict__ Cb,
                    const float* __restrict__ bias, const float* __restrict__ gamma,
                    const float* __restrict__ beta,
                    int M, int N, int K, int flags) {
  const int lane  = threadIdx.x;                    // 0..31 (wave32)
  const int ntile = blockIdx.y * 4 + threadIdx.y;   // 16-column tile
  const int mt    = blockIdx.x * 16;
  const int ksteps = K >> 5;

  int arow = mt + (lane & 15);
  if (arow >= M) arow = M - 1;                      // clamp (stores are guarded)
  const int kb = (lane >> 4) << 4;                  // A frag K offset: 0 or 16
  const __bf16* Arow = A + (long)arow * K;
  const unsigned* Bbase = Bp + ((long)ntile * ksteps) * 256 + lane * 8;

  v8f acc = {};
  for (int ks = 0; ks < ksteps; ++ks) {
    union { v16bf v; uint4 q[2]; } a, b;
    a.q[0] = *(const uint4*)(Arow + ks * 32 + kb);       // K = kb..kb+7
    a.q[1] = *(const uint4*)(Arow + ks * 32 + 8 + kb);   // K = kb+8..kb+15
    const uint4* bp = (const uint4*)(Bbase + ks * 256);
    b.q[0] = bp[0];
    b.q[1] = bp[1];
    acc = __builtin_amdgcn_wmma_f32_16x16x32_bf16(false, a.v, false, b.v,
                                                  (short)0, acc, false, false);
  }

  const int col = ntile * 16 + (lane & 15);
  float bi = (flags & EPI_BIAS) ? bias[col] : 0.0f;
  float gs = 1.0f, bt = 0.0f;
  if (flags & EPI_BN) { gs = gamma[col] * rsqrtf(1.0f + 1e-5f); bt = beta[col]; }
  const int mbase = mt + ((lane >> 4) << 3);        // C layout: m = v + 8*(lane>=16)
#pragma unroll
  for (int v = 0; v < 8; ++v) {
    int row = mbase + v;
    if (row >= M) continue;
    float c = acc[v] + bi;
    if (flags & EPI_RELU) c = fmaxf(c, 0.0f);
    if (flags & EPI_BN)   c = c * gs + bt;
    long o = (long)row * N + col;
    if (Cf) Cf[o] = c;
    if (Cb) Cb[o] = f2bf(c);
  }
}

// ------------------- GAT attention coefficients per node -------------------
// es[n,h] = sum_f h[n,h*32+f]*a_src[h,f]; ed likewise.
__global__ void attn_coef(const float* __restrict__ h, const float* __restrict__ as,
                          const float* __restrict__ ad, float* __restrict__ es,
                          float* __restrict__ ed, int N) {
  int t = blockIdx.x * blockDim.x + threadIdx.x;
  if (t >= N * 4) return;
  int n = t >> 2, hh = t & 3;
  const float* hp = h + (long)n * 128 + hh * 32;
  const float* ap = as + hh * 32;
  const float* bp = ad + hh * 32;
  float s = 0.f, d = 0.f;
#pragma unroll
  for (int f = 0; f < 32; ++f) { float v = hp[f]; s += v * ap[f]; d += v * bp[f]; }
  es[t] = s; ed[t] = d;
}

__global__ void init_seg(float* __restrict__ agg, float* __restrict__ z,
                         unsigned* __restrict__ mkey, int N) {
  int t = blockIdx.x * blockDim.x + threadIdx.x;
  if (t < N * 128) agg[t] = 0.0f;
  if (t < N * 4) { z[t] = 0.0f; mkey[t] = 0u; }   // key 0 == unsigned minimum
}

// pass 1: e = leaky_relu(es[src]+ed[dst]); segment max via uint atomicMax
__global__ void edge_max(const int* __restrict__ ei, int E, int N,
                         const float* __restrict__ es, const float* __restrict__ ed,
                         float* __restrict__ ee, unsigned* __restrict__ mkey) {
  int idx = blockIdx.x * blockDim.x + threadIdx.x;
  if (idx >= E + N) return;
  int s, d;
  if (idx < E) { s = ei[idx]; d = ei[E + idx]; } else { s = idx - E; d = s; }
  float4 a = ((const float4*)es)[s];
  float4 b = ((const float4*)ed)[d];
  float4 e;
  e.x = lrelu(a.x + b.x); e.y = lrelu(a.y + b.y);
  e.z = lrelu(a.z + b.z); e.w = lrelu(a.w + b.w);
  ((float4*)ee)[idx] = e;
  atomicMax(&mkey[d * 4 + 0], fkey(e.x));
  atomicMax(&mkey[d * 4 + 1], fkey(e.y));
  atomicMax(&mkey[d * 4 + 2], fkey(e.z));
  atomicMax(&mkey[d * 4 + 3], fkey(e.w));
}

// pass 2: w = exp(e - m[dst]); z[dst] += w (stores w back over e)
__global__ void edge_exp(const int* __restrict__ ei, int E, int N,
                         float* __restrict__ ee, const unsigned* __restrict__ mkey,
                         float* __restrict__ z) {
  int idx = blockIdx.x * blockDim.x + threadIdx.x;
  if (idx >= E + N) return;
  int d = (idx < E) ? ei[E + idx] : idx - E;
  float4 e = ((const float4*)ee)[idx];
  float4 w;
  w.x = expf(e.x - keyf(mkey[d * 4 + 0]));
  w.y = expf(e.y - keyf(mkey[d * 4 + 1]));
  w.z = expf(e.z - keyf(mkey[d * 4 + 2]));
  w.w = expf(e.w - keyf(mkey[d * 4 + 3]));
  ((float4*)ee)[idx] = w;
  atomicAdd(&z[d * 4 + 0], w.x);
  atomicAdd(&z[d * 4 + 1], w.y);
  atomicAdd(&z[d * 4 + 2], w.z);
  atomicAdd(&z[d * 4 + 3], w.w);
}

// pass 3: agg[dst] += (w/z[dst]) * h[src]   (one thread per (edge, feature))
__global__ void edge_scatter(const int* __restrict__ ei, int E, int N,
                             const float* __restrict__ ee, const float* __restrict__ z,
                             const float* __restrict__ h, float* __restrict__ agg) {
  long t = (long)blockIdx.x * blockDim.x + threadIdx.x;
  long tot = (long)(E + N) * 128;
  if (t >= tot) return;
  int idx = (int)(t >> 7);
  int f = (int)(t & 127);
  int hh = f >> 5;
  int s, d;
  if (idx < E) { s = ei[idx]; d = ei[E + idx]; } else { s = idx - E; d = s; }
  float alpha = ee[idx * 4 + hh] / (z[d * 4 + hh] + 1e-16f);
  atomicAdd(&agg[(long)d * 128 + f], alpha * h[(long)s * 128 + f]);
}

// act = bf16(relu(agg + bias)), C == 128
__global__ void bias_relu_bf16(const float* __restrict__ agg, const float* __restrict__ bias,
                               __bf16* __restrict__ out, int n) {
  int t = blockIdx.x * blockDim.x + threadIdx.x;
  if (t < n) out[t] = f2bf(fmaxf(agg[t] + bias[t & 127], 0.0f));
}

// out[n,f] = mean over 4 heads of agg[n, h*32+f] + b3[f]
__global__ void head_mean(const float* __restrict__ agg, const float* __restrict__ b3,
                          float* __restrict__ out, int N) {
  int t = blockIdx.x * blockDim.x + threadIdx.x;
  if (t >= N * 32) return;
  int n = t >> 5, f = t & 31;
  const float* p = agg + (long)n * 128 + f;
  out[t] = 0.25f * (p[0] + p[32] + p[64] + p[96]) + b3[f];
}

// ---------------------------------------------------------------------------

extern "C" void kernel_launch(void* const* d_in, const int* in_sizes, int n_in,
                              void* d_out, int out_size, void* d_ws, size_t ws_size,
                              hipStream_t stream) {
  const float* x   = (const float*)d_in[0];
  const int*   ei  = (const int*)d_in[1];
  const float* W1  = (const float*)d_in[2];
  const float* a1s = (const float*)d_in[3];
  const float* a1d = (const float*)d_in[4];
  const float* b1  = (const float*)d_in[5];
  const float* W2  = (const float*)d_in[6];
  const float* a2s = (const float*)d_in[7];
  const float* a2d = (const float*)d_in[8];
  const float* b2  = (const float*)d_in[9];
  const float* M1w = (const float*)d_in[10];
  const float* M1b = (const float*)d_in[11];
  const float* g1  = (const float*)d_in[12];
  const float* be1 = (const float*)d_in[13];
  const float* M2w = (const float*)d_in[14];
  const float* M2b = (const float*)d_in[15];
  const float* g2  = (const float*)d_in[16];
  const float* be2 = (const float*)d_in[17];
  const float* W3  = (const float*)d_in[18];
  const float* a3s = (const float*)d_in[19];
  const float* a3d = (const float*)d_in[20];
  const float* b3  = (const float*)d_in[21];
  float* out = (float*)d_out;

  const int N  = in_sizes[0] / 128;
  const int E  = in_sizes[1] / 2;
  const int Et = E + N;

  // workspace carve (256B aligned)
  char* p = (char*)d_ws;
  auto carve = [&](size_t bytes) -> void* {
    void* r = (void*)p; p += (bytes + 255) & ~(size_t)255; return r;
  };
  __bf16*   bfA  = (__bf16*)carve((size_t)N * 128 * sizeof(__bf16));
  __bf16*   bfB  = (__bf16*)carve((size_t)N * 64  * sizeof(__bf16));
  unsigned* pW1  = (unsigned*)carve(128 * 128 * 2);
  unsigned* pW2  = (unsigned*)carve(128 * 128 * 2);
  unsigned* pW3  = (unsigned*)carve(128 * 128 * 2);
  unsigned* pM1  = (unsigned*)carve(128 * 64 * 2);
  unsigned* pM2  = (unsigned*)carve(64 * 128 * 2);
  float*    h    = (float*)carve((size_t)N * 128 * sizeof(float));
  float*    agg  = (float*)carve((size_t)N * 128 * sizeof(float));
  float*    es   = (float*)carve((size_t)N * 4 * sizeof(float));
  float*    ed   = (float*)carve((size_t)N * 4 * sizeof(float));
  float*    ee   = (float*)carve((size_t)Et * 4 * sizeof(float));
  unsigned* mkey = (unsigned*)carve((size_t)N * 4 * sizeof(unsigned));
  float*    z    = (float*)carve((size_t)N * 4 * sizeof(float));

  const int B256 = 256;
  auto blk = [&](long n) { return (unsigned)((n + B256 - 1) / B256); };

  // ---- pack weights into WMMA B-fragments; cvt input to bf16 ----
  pack_wmma_b<<<blk(128 * 128 / 2), B256, 0, stream>>>(W1, pW1, 128, 128);
  pack_wmma_b<<<blk(128 * 128 / 2), B256, 0, stream>>>(W2, pW2, 128, 128);
  pack_wmma_b<<<blk(128 * 128 / 2), B256, 0, stream>>>(W3, pW3, 128, 128);
  pack_wmma_b<<<blk(128 * 64 / 2),  B256, 0, stream>>>(M1w, pM1, 128, 64);
  pack_wmma_b<<<blk(64 * 128 / 2),  B256, 0, stream>>>(M2w, pM2, 64, 128);
  cvt_f32_bf16<<<blk((long)N * 128), B256, 0, stream>>>(x, bfA, N * 128);

  dim3 gblk(32, 4);
  dim3 g128((N + 15) / 16, 128 / 64);
  dim3 g64 ((N + 15) / 16, 64 / 64);

  auto gat_layer = [&](const unsigned* pW, const float* as, const float* ad,
                       const float* bias, bool last) {
    gemm_wmma_bf16<<<g128, gblk, 0, stream>>>(bfA, pW, h, nullptr,
                                              nullptr, nullptr, nullptr,
                                              N, 128, 128, 0);
    attn_coef<<<blk((long)N * 4), B256, 0, stream>>>(h, as, ad, es, ed, N);
    init_seg<<<blk((long)N * 128), B256, 0, stream>>>(agg, z, mkey, N);
    edge_max<<<blk(Et), B256, 0, stream>>>(ei, E, N, es, ed, ee, mkey);
    edge_exp<<<blk(Et), B256, 0, stream>>>(ei, E, N, ee, mkey, z);
    edge_scatter<<<blk((long)Et * 128), B256, 0, stream>>>(ei, E, N, ee, z, h, agg);
    if (!last)
      bias_relu_bf16<<<blk((long)N * 128), B256, 0, stream>>>(agg, bias, bfA, N * 128);
  };

  // conv0 -> relu -> bf16
  gat_layer(pW1, a1s, a1d, b1, false);

  // MLP: (HF->64, bias+relu+bn) -> (64->HF, bias+relu+bn), fused epilogues
  gemm_wmma_bf16<<<g64, gblk, 0, stream>>>(bfA, pM1, nullptr, bfB,
                                           M1b, g1, be1, N, 64, 128,
                                           EPI_BIAS | EPI_RELU | EPI_BN);
  gemm_wmma_bf16<<<g128, gblk, 0, stream>>>(bfB, pM2, nullptr, bfA,
                                            M2b, g2, be2, N, 128, 64,
                                            EPI_BIAS | EPI_RELU | EPI_BN);

  // conv1 -> relu -> bf16
  gat_layer(pW2, a2s, a2d, b2, false);

  // conv2 (concat=False): aggregate then mean over heads + b3
  gat_layer(pW3, a3s, a3d, nullptr, true);
  head_mean<<<blk((long)N * 32), B256, 0, stream>>>(agg, b3, out, N);
}
